// BigramSelfAttentionLanguageModel_24120536335173
// MI455X (gfx1250) — matrix operand order repaired and checked by reading, not compiled, not run on hardware
//
#include <hip/hip_runtime.h>

// ---------------------------------------------------------------------------
// Types / WMMA helpers (gfx1250, wave32)
// ---------------------------------------------------------------------------
typedef __attribute__((ext_vector_type(16))) __bf16 v16bf;
typedef __attribute__((ext_vector_type(8)))  float  v8f;

#define DEVINL __device__ __forceinline__

DEVINL v8f vzero8() {
  v8f z;
#pragma unroll
  for (int i = 0; i < 8; ++i) z[i] = 0.0f;
  return z;
}

DEVINL v8f wmma_bf16(v16bf a, v16bf b, v8f c) {
  // D = A(16x32 bf16) * B(32x16 bf16) + C(16x16 f32)
  return __builtin_amdgcn_wmma_f32_16x16x32_bf16(
      /*neg_a=*/false, a, /*neg_b=*/false, b,
      /*c_mod=*/(short)0, c, /*reuse_a=*/false, /*reuse_b=*/false);
}

union Frag { uint4 u[2]; v16bf v; };
union BfBits { __bf16 h; unsigned short s; };

// A-matrix fragment (16x32 bf16, row-major source, leading dim ld elements).
// Lane layout: row = lane&15 ; K = kb + e (e<8), kb+16 + (e-8) (e>=8), kb=(lane>>4)*8
DEVINL v16bf load_frag_a(const __bf16* __restrict__ base, int ld) {
  int lane = threadIdx.x & 31;
  const __bf16* p = base + (size_t)(lane & 15) * ld + ((lane >> 4) << 3);
  Frag f;
  f.u[0] = *(const uint4*)(p);
  f.u[1] = *(const uint4*)(p + 16);
  return f.v;
}

// B-matrix fragment (32x16 bf16) from a [N][K]-layout (K contiguous) source.
// Lane layout: col = lane&15 ; K = kb + e (e<8), kb+16 + (e-8) (e>=8), kb=(lane>>4)*8
DEVINL v16bf load_frag_b(const __bf16* __restrict__ baseT, int ld) {
  int lane = threadIdx.x & 31;
  const __bf16* p = baseT + (size_t)(lane & 15) * ld + ((lane >> 4) << 3);
  Frag f;
  f.u[0] = *(const uint4*)(p);
  f.u[1] = *(const uint4*)(p + 16);
  return f.v;
}

// Async 16-byte global -> LDS copy (CDNA5 GLOBAL_LOAD_ASYNC_TO_LDS_B128, GV mode).
// Tracked by ASYNCcnt; must s_wait_asynccnt before the barrier that publishes LDS.
DEVINL void async_copy_b128(unsigned lds_addr, const void* gptr) {
  unsigned long long ga = (unsigned long long)(size_t)gptr;
  asm volatile("global_load_async_to_lds_b128 %0, %1, off"
               :: "v"(lds_addr), "v"(ga)
               : "memory");
}

DEVINL void wait_asynccnt0() {
  asm volatile("s_wait_asynccnt 0x0" ::: "memory");
}

// ---------------------------------------------------------------------------
// f32 -> bf16 convert (8 elements / thread, b128 in/out; n % 8 == 0)
// ---------------------------------------------------------------------------
__global__ __launch_bounds__(256) void f2bf_kernel(const float* __restrict__ in,
                                                   __bf16* __restrict__ out, int n) {
  int i8 = (blockIdx.x * 256 + threadIdx.x) * 8;
  if (i8 >= n) return;
  float4 a = *(const float4*)(in + i8);
  float4 b = *(const float4*)(in + i8 + 4);
  union { unsigned short s[8]; uint4 u; } pk;
  BfBits t;
  t.h = (__bf16)a.x; pk.s[0] = t.s;
  t.h = (__bf16)a.y; pk.s[1] = t.s;
  t.h = (__bf16)a.z; pk.s[2] = t.s;
  t.h = (__bf16)a.w; pk.s[3] = t.s;
  t.h = (__bf16)b.x; pk.s[4] = t.s;
  t.h = (__bf16)b.y; pk.s[5] = t.s;
  t.h = (__bf16)b.z; pk.s[6] = t.s;
  t.h = (__bf16)b.w; pk.s[7] = t.s;
  *(uint4*)(out + i8) = pk.u;
}

// ---------------------------------------------------------------------------
// Embedding: x[row][c] = tok_emb[idx[row]][c] + pos_emb[row%T][c]  (bf16 out)
// One block per row; each thread handles 4 channels (C = 1024 = 256*4).
// ---------------------------------------------------------------------------
__global__ __launch_bounds__(256) void embed_kernel(const int* __restrict__ idx,
                                                    const float* __restrict__ tok,
                                                    const float* __restrict__ pos,
                                                    __bf16* __restrict__ x,
                                                    int T, int C) {
  int row = blockIdx.x;
  int t = row % T;
  int token = idx[row];
  const float* te = tok + (size_t)token * C;
  const float* pe = pos + (size_t)t * C;
  __bf16* xo = x + (size_t)row * C;
  int c = threadIdx.x * 4;
  float4 a = *(const float4*)(te + c);
  float4 b = *(const float4*)(pe + c);
  union { unsigned short s[4]; uint2 u; } pk;
  BfBits w;
  w.h = (__bf16)(a.x + b.x); pk.s[0] = w.s;
  w.h = (__bf16)(a.y + b.y); pk.s[1] = w.s;
  w.h = (__bf16)(a.z + b.z); pk.s[2] = w.s;
  w.h = (__bf16)(a.w + b.w); pk.s[3] = w.s;
  *(uint2*)(xo + c) = pk.u;
}

// ---------------------------------------------------------------------------
// Tiled WMMA GEMM:  C[M,N] = A[M,1024] * B[1024,N] (+bias)
// Block tile 64(M) x 256(N), 8 waves as 2x4, wave tile 32x64, K-step 32.
// Double-buffered: async A copy + global B fetch for tile i+1 overlap the
// 8 WMMAs of tile i; one barrier per K-step.
// ---------------------------------------------------------------------------
template <bool OUT_F32, bool BIAS>
__global__ __launch_bounds__(256) void gemm_bf16(const __bf16* __restrict__ A,
                                                 const __bf16* __restrict__ Bm,
                                                 const float* __restrict__ bias,
                                                 float* __restrict__ Cf,
                                                 __bf16* __restrict__ Cb,
                                                 int M, int N) {
  constexpr int K = 1024, BK = 32, NSTEP = K / BK;
  __shared__ __align__(16) __bf16 As[2][64][BK];   // 8 KB
  __shared__ __align__(16) __bf16 Bs[2][256][BK];  // 32 KB (transposed [n][k])

  const int tid  = threadIdx.x;
  const int wave = tid >> 5;
  const int lane = tid & 31;
  const int m0 = blockIdx.x * 64;
  const int n0 = blockIdx.y * 256;
  const int wml = (wave & 1) * 32;   // wave rows (local)
  const int wnl = (wave >> 1) * 64;  // wave cols (local)

  v8f acc[2][4];
#pragma unroll
  for (int t = 0; t < 2; ++t)
#pragma unroll
    for (int j = 0; j < 4; ++j) acc[t][j] = vzero8();

  // staging task assignments (fixed per thread)
  const int ar  = tid >> 2;        // A: row 0..63
  const int ac8 = (tid & 3) * 8;   // A: k-chunk 0/8/16/24
  const int kg  = tid >> 5;        // B: 8 groups of 4 K-rows
  const int nc  = tid & 31;        // B: 32 chunks of 8 N-cols
  const unsigned aLds[2] = { (unsigned)(size_t)&As[0][ar][ac8],
                             (unsigned)(size_t)&As[1][ar][ac8] };
  const __bf16* aRow = A + (size_t)(m0 + ar) * K + ac8;
  const __bf16* bRow = Bm + (size_t)(kg * 4) * N + (n0 + nc * 8);

  // prologue: start tile 0 in flight
  async_copy_b128(aLds[0], aRow);
  uint4 r0 = *(const uint4*)(bRow);
  uint4 r1 = *(const uint4*)(bRow + N);
  uint4 r2 = *(const uint4*)(bRow + 2 * (size_t)N);
  uint4 r3 = *(const uint4*)(bRow + 3 * (size_t)N);

  for (int i = 0; i < NSTEP; ++i) {
    const int cur = i & 1;

    // commit staged B regs into LDS buffer `cur` (transposed [n][k])
    {
      const unsigned short* s0 = (const unsigned short*)&r0;
      const unsigned short* s1 = (const unsigned short*)&r1;
      const unsigned short* s2 = (const unsigned short*)&r2;
      const unsigned short* s3 = (const unsigned short*)&r3;
#pragma unroll
      for (int c8 = 0; c8 < 8; ++c8) {
        uint2 val;
        val.x = (unsigned)s0[c8] | ((unsigned)s1[c8] << 16);
        val.y = (unsigned)s2[c8] | ((unsigned)s3[c8] << 16);
        *(uint2*)&Bs[cur][nc * 8 + c8][kg * 4] = val;
      }
    }
    wait_asynccnt0();   // async A writes for tile i complete
    __syncthreads();    // publish buffer `cur`

    // start tile i+1 into buffer 1-cur (overlaps compute below)
    if (i + 1 < NSTEP) {
      const size_t koff = (size_t)(i + 1) * BK;
      async_copy_b128(aLds[cur ^ 1], aRow + koff);
      const __bf16* gp = bRow + koff * N;
      r0 = *(const uint4*)(gp);
      r1 = *(const uint4*)(gp + N);
      r2 = *(const uint4*)(gp + 2 * (size_t)N);
      r3 = *(const uint4*)(gp + 3 * (size_t)N);
      if (i + 2 < NSTEP)  // prefetch tile i+2 -> global_prefetch_b8
        __builtin_prefetch(bRow + (size_t)(i + 2) * BK * N, 0, 1);
    }

    // compute from buffer `cur`
    v16bf a0 = load_frag_a(&As[cur][wml][0], BK);
    v16bf a1 = load_frag_a(&As[cur][wml + 16][0], BK);
#pragma unroll
    for (int j = 0; j < 4; ++j) {
      v16bf b = load_frag_b(&Bs[cur][wnl + j * 16][0], BK);
      acc[0][j] = wmma_bf16(a0, b, acc[0][j]);
      acc[1][j] = wmma_bf16(a1, b, acc[1][j]);
    }
    // no trailing barrier: next iteration writes only buffer 1-cur, whose
    // previous readers are fenced by this iteration's barrier.
  }

  // epilogue: C/D layout -> lane col = lane&15, VGPR r -> row r + 8*(lane>>4)
  const int hi = (lane >> 4) << 3;
#pragma unroll
  for (int t = 0; t < 2; ++t)
#pragma unroll
    for (int j = 0; j < 4; ++j) {
      int col = n0 + wnl + j * 16 + (lane & 15);
      float bv = BIAS ? bias[col] : 0.0f;
#pragma unroll
      for (int r = 0; r < 8; ++r) {
        int row = m0 + wml + t * 16 + hi + r;
        float v = acc[t][j][r] + bv;
        if (OUT_F32) Cf[(size_t)row * N + col] = v;
        else         Cb[(size_t)row * N + col] = (__bf16)v;
      }
    }
}

// ---------------------------------------------------------------------------
// Flash attention: one block per 16-query tile, 8 waves split C=1024.
// Scores via WMMA (q@k^T), online softmax by wave 0, P@V via WMMA.
// q fragments are loop-invariant and held in VGPRs for the whole kernel.
// ---------------------------------------------------------------------------
__global__ __launch_bounds__(256) void attention_kernel(const __bf16* __restrict__ q,
                                                        const __bf16* __restrict__ k,
                                                        const __bf16* __restrict__ v,
                                                        __bf16* __restrict__ out) {
  constexpr int T = 2048, C = 1024;
  __shared__ __align__(16) __bf16 vT[C][32];   // 64 KB, [c][key]
  __shared__ float  S[16][32];                 // score tile (f32 partial sums)
  __shared__ __align__(16) __bf16 P[16][32];   // softmaxed probs (bf16)
  __shared__ float  mrow[16], lrow[16], facrow[16];

  const int tid  = threadIdx.x;
  const int wave = tid >> 5;
  const int lane = tid & 31;
  const int tilesPerB = T / 16;
  const int b  = blockIdx.x / tilesPerB;
  const int qt = (blockIdx.x % tilesPerB) * 16;   // local query base
  const size_t rowbase = (size_t)b * T;
  const int wc = wave * 128;                      // this wave's C slice
  const int hi = (lane >> 4) << 3;
  const int col = lane & 15;

  if (tid < 16) { mrow[tid] = -1e30f; lrow[tid] = 0.0f; }

  v8f acc[8];
#pragma unroll
  for (int g = 0; g < 8; ++g) acc[g] = vzero8();

  // hoist loop-invariant q fragments (this wave's 4 K-steps of its C slice)
  v16bf aqf[4];
#pragma unroll
  for (int s = 0; s < 4; ++s)
    aqf[s] = load_frag_a(q + (rowbase + qt) * (size_t)C + wc + s * 32, C);

  const int nkb = (qt + 16 + 31) >> 5;  // causal: key blocks of 32 needed

  for (int kbi = 0; kbi < nkb; ++kbi) {
    const int kb = kbi * 32;

    // phase A: zero S, stage v^T tile [1024][32] into LDS (4 tasks/thread)
    for (int i = tid; i < 16 * 32; i += 256) ((float*)S)[i] = 0.0f;
    for (int task = tid; task < 1024; task += 256) {
      int kg = task & 7;    // group of 4 keys
      int cc = task >> 3;   // chunk of 8 channels
      const __bf16* gp = v + (rowbase + kb + kg * 4) * (size_t)C + cc * 8;
      uint4 r0 = *(const uint4*)(gp);
      uint4 r1 = *(const uint4*)(gp + C);
      uint4 r2 = *(const uint4*)(gp + 2 * C);
      uint4 r3 = *(const uint4*)(gp + 3 * C);
      const unsigned short* s0 = (const unsigned short*)&r0;
      const unsigned short* s1 = (const unsigned short*)&r1;
      const unsigned short* s2 = (const unsigned short*)&r2;
      const unsigned short* s3 = (const unsigned short*)&r3;
#pragma unroll
      for (int c8 = 0; c8 < 8; ++c8) {
        uint2 val;
        val.x = (unsigned)s0[c8] | ((unsigned)s1[c8] << 16);
        val.y = (unsigned)s2[c8] | ((unsigned)s3[c8] << 16);
        *(uint2*)&vT[cc * 8 + c8][kg * 4] = val;
      }
    }
    __syncthreads();

    // phase B: partial scores over this wave's C slice (4 K-steps, 2 key tiles)
    v8f sc0 = vzero8(), sc1 = vzero8();
#pragma unroll
    for (int s = 0; s < 4; ++s) {
      const int cb = wc + s * 32;
      v16bf b0 = load_frag_b(k + (rowbase + kb) * (size_t)C + cb, C);       // keys kb..kb+15
      v16bf b1 = load_frag_b(k + (rowbase + kb + 16) * (size_t)C + cb, C);  // keys kb+16..kb+31
      sc0 = wmma_bf16(aqf[s], b0, sc0);
      sc1 = wmma_bf16(aqf[s], b1, sc1);
    }
#pragma unroll
    for (int r = 0; r < 8; ++r) {
      atomicAdd(&S[hi + r][col], sc0[r]);
      atomicAdd(&S[hi + r][col + 16], sc1[r]);
    }
    __syncthreads();

    // phase C: online softmax for this key block (lanes 0..15 of wave 0)
    if (tid < 16) {
      const int rr = tid;
      const int qg = qt + rr;
      const float scale = 0.03125f;  // 1024^-0.5
      float m_old = mrow[rr];
      float mx = m_old;
      float sv[32];
#pragma unroll
      for (int j = 0; j < 32; ++j) {
        float x = (kb + j <= qg) ? S[rr][j] * scale : -1e30f;
        sv[j] = x;
        mx = fmaxf(mx, x);
      }
      float sum = 0.0f;
#pragma unroll
      for (int j = 0; j < 32; ++j) {
        float p = (kb + j <= qg) ? __expf(sv[j] - mx) : 0.0f;
        P[rr][j] = (__bf16)p;
        sum += p;
      }
      float fac = __expf(m_old - mx);
      facrow[rr] = fac;
      lrow[rr] = lrow[rr] * fac + sum;
      mrow[rr] = mx;
    }
    __syncthreads();

    // phase D: rescale accumulators and add P @ V for this wave's C slice
    float fac[8];
#pragma unroll
    for (int r = 0; r < 8; ++r) fac[r] = facrow[hi + r];
    v16bf ap = load_frag_a(&P[0][0], 32);
#pragma unroll
    for (int g = 0; g < 8; ++g) {
      v16bf bv = load_frag_b(&vT[wc + g * 16][0], 32);
#pragma unroll
      for (int r = 0; r < 8; ++r) acc[g][r] *= fac[r];
      acc[g] = wmma_bf16(ap, bv, acc[g]);
    }
    __syncthreads();
  }

  // epilogue: out = acc / l
  float li[8];
#pragma unroll
  for (int r = 0; r < 8; ++r) li[r] = lrow[hi + r];
#pragma unroll
  for (int g = 0; g < 8; ++g) {
#pragma unroll
    for (int r = 0; r < 8; ++r) {
      int row = qt + hi + r;
      int c = wc + g * 16 + col;
      out[(rowbase + row) * (size_t)C + c] = (__bf16)(acc[g][r] / li[r]);
    }
  }
}

// ---------------------------------------------------------------------------
// Loss: single-pass online logsumexp per row (float4 loads), then
// deterministic tree reductions.
// ---------------------------------------------------------------------------
__global__ __launch_bounds__(256) void row_loss_kernel(const float* __restrict__ logits,
                                                       const int* __restrict__ tgt,
                                                       float* __restrict__ rowloss, int V) {
  const int row = blockIdx.x;
  const float* p = logits + (size_t)row * V;
  const int tid = threadIdx.x;

  float m = -1e30f, s = 0.0f;
  for (int i = tid * 4; i < V; i += 1024) {
    float4 x = *(const float4*)(p + i);
    float xm = fmaxf(fmaxf(x.x, x.y), fmaxf(x.z, x.w));
    float mn = fmaxf(m, xm);
    s = s * __expf(m - mn) + __expf(x.x - mn) + __expf(x.y - mn) +
        __expf(x.z - mn) + __expf(x.w - mn);
    m = mn;
  }

  __shared__ float ms[256], ss[256];
  ms[tid] = m; ss[tid] = s;
  __syncthreads();
  for (int st = 128; st > 0; st >>= 1) {
    if (tid < st) {
      float m2 = ms[tid + st], s2 = ss[tid + st];
      float mn = fmaxf(ms[tid], m2);
      ss[tid] = ss[tid] * __expf(ms[tid] - mn) + s2 * __expf(m2 - mn);
      ms[tid] = mn;
    }
    __syncthreads();
  }
  if (tid == 0) rowloss[row] = ms[0] + __logf(ss[0]) - p[tgt[row]];
}

__global__ __launch_bounds__(256) void final_loss_kernel(const float* __restrict__ rowloss,
                                                         float* __restrict__ out, int n) {
  __shared__ float red[256];
  const int tid = threadIdx.x;
  float s = 0.0f;
  for (int i = tid; i < n; i += 256) s += rowloss[i];
  red[tid] = s;
  __syncthreads();
  for (int st = 128; st > 0; st >>= 1) {
    if (tid < st) red[tid] += red[tid + st];
    __syncthreads();
  }
  if (tid == 0) *out = red[0] / (float)n;
}

// ---------------------------------------------------------------------------
// Host launcher
// ---------------------------------------------------------------------------
extern "C" void kernel_launch(void* const* d_in, const int* in_sizes, int n_in,
                              void* d_out, int out_size, void* d_ws, size_t ws_size,
                              hipStream_t stream) {
  (void)in_sizes; (void)n_in; (void)ws_size;
  constexpr int Bn = 4, T = 2048, C = 1024, V = 32000;
  constexpr int M = Bn * T;  // 8192

  const int*   idx     = (const int*)  d_in[0];
  const int*   targets = (const int*)  d_in[1];
  const float* tok_emb = (const float*)d_in[2];
  const float* pos_emb = (const float*)d_in[3];
  const float* Wk      = (const float*)d_in[4];
  const float* Wq      = (const float*)d_in[5];
  const float* Wv      = (const float*)d_in[6];
  const float* Wlm     = (const float*)d_in[7];
  const float* blm     = (const float*)d_in[8];

  char* cur = (char*)d_ws;
  auto take = [&](size_t bytes) {
    char* p = cur;
    cur += (bytes + 255) & ~(size_t)255;
    return p;
  };
  __bf16* xb      = (__bf16*)take((size_t)M * C * 2);
  __bf16* wqb     = (__bf16*)take((size_t)C * C * 2);
  __bf16* wkb     = (__bf16*)take((size_t)C * C * 2);
  __bf16* wvb     = (__bf16*)take((size_t)C * C * 2);
  __bf16* wlmb    = (__bf16*)take((size_t)C * V * 2);
  __bf16* qb      = (__bf16*)take((size_t)M * C * 2);
  __bf16* kbuf    = (__bf16*)take((size_t)M * C * 2);
  __bf16* vbuf    = (__bf16*)take((size_t)M * C * 2);
  __bf16* ob      = (__bf16*)take((size_t)M * C * 2);
  float*  rowloss = (float*) take((size_t)M * 4);

  float* logits = (float*)d_out;
  float* lossp  = logits + (size_t)(out_size - 1);

  // 1) weight converts (f32 -> bf16), 8 elems/thread
  f2bf_kernel<<<(C * C / 8 + 255) / 256, 256, 0, stream>>>(Wq, wqb, C * C);
  f2bf_kernel<<<(C * C / 8 + 255) / 256, 256, 0, stream>>>(Wk, wkb, C * C);
  f2bf_kernel<<<(C * C / 8 + 255) / 256, 256, 0, stream>>>(Wv, wvb, C * C);
  f2bf_kernel<<<(C * V / 8 + 255) / 256, 256, 0, stream>>>(Wlm, wlmb, C * V);

  // 2) embedding
  embed_kernel<<<M, 256, 0, stream>>>(idx, tok_emb, pos_emb, xb, T, C);

  // 3) q,k,v = x @ W  (M=8192, N=1024, K=1024)
  dim3 g1(M / 64, C / 256);
  gemm_bf16<false, false><<<g1, 256, 0, stream>>>(xb, wqb, nullptr, nullptr, qb, M, C);
  gemm_bf16<false, false><<<g1, 256, 0, stream>>>(xb, wkb, nullptr, nullptr, kbuf, M, C);
  gemm_bf16<false, false><<<g1, 256, 0, stream>>>(xb, wvb, nullptr, nullptr, vbuf, M, C);

  // 4) causal flash attention
  attention_kernel<<<M / 16, 256, 0, stream>>>(qb, kbuf, vbuf, ob);

  // 5) logits = attn_out @ Wlm + blm  (f32 straight into d_out)
  dim3 g2(M / 64, V / 256);
  gemm_bf16<true, true><<<g2, 256, 0, stream>>>(ob, wlmb, blm, logits, nullptr, M, V);

  // 6) loss (single pass over logits)
  row_loss_kernel<<<M, 256, 0, stream>>>(logits, targets, rowloss, V);
  final_loss_kernel<<<1, 256, 0, stream>>>(rowloss, lossp, M);
}
